// ActionEncoder_20839181320498
// MI455X (gfx1250) — matrix operand order repaired and checked by reading, not compiled, not run on hardware
//
#include <hip/hip_runtime.h>
#include <hip/hip_bf16.h>

typedef __attribute__((ext_vector_type(16))) __bf16 v16bf;
typedef __attribute__((ext_vector_type(8)))  float  v8f;

union FragBF {
    unsigned short s[16];
    uint4          q[2];
    v16bf          v;
};

__device__ __forceinline__ unsigned short f2bf(float f) {
    unsigned int u = __float_as_uint(f);
    u += 0x7fffu + ((u >> 16) & 1u);        // round-to-nearest-even
    return (unsigned short)(u >> 16);
}

// ---------------------------------------------------------------------------
// Fused gather + 2-layer MLP:  out = leaky_relu(X @ W1 + b1) @ W2 + b2
// X rows are gathered concatenations of K1/256 embedding segments.
// W1t: (512 x K1) bf16 (pre-transposed), W2t: (256 x 512) bf16.
// BFT = true: tables pre-converted to bf16 (direct fragment loads).
// Block = 256 threads = 8 waves; 64 output rows per block.
// ---------------------------------------------------------------------------
template <int K1, bool BFT>
__global__ __launch_bounds__(256)
void mlp2_wmma_kernel(const void* __restrict__ tab0, const int* __restrict__ idx0,
                      const void* __restrict__ tab1, const int* __restrict__ idx1,
                      const void* __restrict__ tab2, const int* __restrict__ idx2,
                      const void* __restrict__ tab3, const int* __restrict__ idx3,
                      const unsigned short* __restrict__ w1t,
                      const float* __restrict__ b1,
                      const unsigned short* __restrict__ w2t,
                      const float* __restrict__ b2,
                      float* __restrict__ out,
                      long long outRowBase, int nRows)
{
    __shared__ unsigned short Hlds[64 * 512];   // hidden layer, bf16, 64 KB

    const int tid  = threadIdx.x;
    const int wave = tid >> 5;
    const int lane = tid & 31;
    const int lr   = lane & 15;     // lane-in-halfwave
    const int half = lane >> 4;     // 0 or 1
    const int rt   = wave & 3;      // row-tile (4 x 16 rows)
    const int cg   = wave >> 2;     // col group (2 x 256 cols for L1)

    const int wgRow0 = blockIdx.x * 64;
    const int rowG   = wgRow0 + rt * 16 + lr;
    const int rowGc  = min(rowG, nRows - 1);

    // gather indices for this lane's row (one embedding row per 256-wide segment)
    const int sr0 = idx0[rowGc];
    const int sr1 = idx1[rowGc];
    const int sr2 = (K1 > 512) ? idx2[rowGc] : 0;
    const int sr3 = (K1 > 512) ? idx3[rowGc] : 0;

    // ---------------- layer 1: 16 accumulator tiles (16 rows x 256 cols/wave)
    v8f acc[16];
#pragma unroll
    for (int j = 0; j < 16; ++j)
        acc[j] = (v8f){0.f, 0.f, 0.f, 0.f, 0.f, 0.f, 0.f, 0.f};

#pragma unroll 4
    for (int k0 = 0; k0 < K1; k0 += 32) {
        const int seg    = k0 >> 8;
        const void* tp   = (seg == 0) ? tab0 : (seg == 1) ? tab1 : (seg == 2) ? tab2 : tab3;
        const int    sr  = (seg == 0) ? sr0  : (seg == 1) ? sr1  : (seg == 2) ? sr2  : sr3;
        const int    koff = k0 & 255;

        // A fragment (16x32 bf16): lane(half,lr) holds row lr,
        // K = k0 + half*8 + [0..7]  and  K = k0 + 16 + half*8 + [0..7]
        FragBF a;
        if (BFT) {
            const unsigned short* ap = (const unsigned short*)tp + (long)sr * 256 + koff + half * 8;
            a.q[0] = *(const uint4*)(ap);
            a.q[1] = *(const uint4*)(ap + 16);
        } else {
            const float* ap = (const float*)tp + (long)sr * 256 + koff + half * 8;
            const float4 g0 = *(const float4*)(ap + 0);
            const float4 g1 = *(const float4*)(ap + 4);
            const float4 g2 = *(const float4*)(ap + 16);
            const float4 g3 = *(const float4*)(ap + 20);
            a.s[0]  = f2bf(g0.x); a.s[1]  = f2bf(g0.y); a.s[2]  = f2bf(g0.z); a.s[3]  = f2bf(g0.w);
            a.s[4]  = f2bf(g1.x); a.s[5]  = f2bf(g1.y); a.s[6]  = f2bf(g1.z); a.s[7]  = f2bf(g1.w);
            a.s[8]  = f2bf(g2.x); a.s[9]  = f2bf(g2.y); a.s[10] = f2bf(g2.z); a.s[11] = f2bf(g2.w);
            a.s[12] = f2bf(g3.x); a.s[13] = f2bf(g3.y); a.s[14] = f2bf(g3.z); a.s[15] = f2bf(g3.w);
        }

#pragma unroll
        for (int j = 0; j < 16; ++j) {
            // B fragment (32x16 bf16): lane(half,lr) holds col lr, K = k0 + half*8 + [0..7] and k0 + 16 + half*8 + [0..7]
            const int ncol = cg * 256 + j * 16 + lr;
            const unsigned short* bp = w1t + (long)ncol * K1 + k0 + half * 8;
            FragBF b;
            b.q[0] = *(const uint4*)(bp);
            b.q[1] = *(const uint4*)(bp + 16);
            acc[j] = __builtin_amdgcn_wmma_f32_16x16x32_bf16(
                false, a.v, false, b.v, (short)0, acc[j], false, false);
        }
    }

    // bias + leaky-relu -> H in LDS (bf16, row-major, ld = 512)
#pragma unroll
    for (int j = 0; j < 16; ++j) {
        const int ncol = cg * 256 + j * 16 + lr;
        const float bv = b1[ncol];
#pragma unroll
        for (int v = 0; v < 8; ++v) {
            float x = acc[j][v] + bv;                  // D row = v + 8*half
            x = (x > 0.f) ? x : 0.01f * x;             // LeakyReLU(0.01)
            Hlds[(rt * 16 + v + 8 * half) * 512 + ncol] = f2bf(x);
        }
    }
    __syncthreads();

    // ---------------- layer 2: 8 accumulator tiles (16 rows x 128 cols/wave)
    v8f acc2[8];
#pragma unroll
    for (int j = 0; j < 8; ++j)
        acc2[j] = (v8f){0.f, 0.f, 0.f, 0.f, 0.f, 0.f, 0.f, 0.f};

#pragma unroll 4
    for (int k0 = 0; k0 < 512; k0 += 32) {
        const unsigned short* ap = &Hlds[(rt * 16 + lr) * 512 + k0 + half * 8];
        FragBF a;
        a.q[0] = *(const uint4*)(ap);        // K = k0 + half*8 + [0..7]
        a.q[1] = *(const uint4*)(ap + 16);   // K = k0 + 16 + half*8 + [0..7]
#pragma unroll
        for (int j = 0; j < 8; ++j) {
            const int ncol = cg * 128 + j * 16 + lr;
            const unsigned short* bp = w2t + (long)ncol * 512 + k0 + half * 8;
            FragBF b;
            b.q[0] = *(const uint4*)(bp);
            b.q[1] = *(const uint4*)(bp + 16);
            acc2[j] = __builtin_amdgcn_wmma_f32_16x16x32_bf16(
                false, a.v, false, b.v, (short)0, acc2[j], false, false);
        }
    }

    // bias + store fp32 output
#pragma unroll
    for (int j = 0; j < 8; ++j) {
        const int ncol = cg * 128 + j * 16 + lr;
        const float bv = b2[ncol];
#pragma unroll
        for (int v = 0; v < 8; ++v) {
            const int r = wgRow0 + rt * 16 + v + 8 * half;
            if (r < nRows)
                out[(outRowBase + r) * 256 + ncol] = acc2[j][v] + bv;
        }
    }
}

// in: (K x N) f32  ->  out: (N x K) bf16   (transpose + downconvert, one-time)
__global__ void cvt_transpose_kernel(const float* __restrict__ in,
                                     unsigned short* __restrict__ out,
                                     int K, int N)
{
    const int i = blockIdx.x * blockDim.x + threadIdx.x;
    if (i >= K * N) return;
    const int n = i / K;
    const int k = i - n * K;
    out[(long)n * K + k] = f2bf(in[(long)k * N + n]);
}

// elementwise f32 -> bf16 table conversion, 4 elems/thread
__global__ void cvt_table_kernel(const float* __restrict__ in,
                                 unsigned short* __restrict__ out, int n4)
{
    const int i = blockIdx.x * blockDim.x + threadIdx.x;
    if (i >= n4) return;
    const float4 f = ((const float4*)in)[i];
    ushort4 o;
    o.x = f2bf(f.x); o.y = f2bf(f.y); o.z = f2bf(f.z); o.w = f2bf(f.w);
    ((ushort4*)out)[i] = o;
}

// broadcast wait_emb into the first n_wait output rows
__global__ void wait_fill_kernel(const float* __restrict__ wait_emb,
                                 float* __restrict__ out, int n)
{
    const int i = blockIdx.x * blockDim.x + threadIdx.x;
    if (i < n) out[i] = wait_emb[i & 255];
}

extern "C" void kernel_launch(void* const* d_in, const int* in_sizes, int n_in,
                              void* d_out, int out_size, void* d_ws, size_t ws_size,
                              hipStream_t stream)
{
    (void)n_in;
    const float* op_emb   = (const float*)d_in[0];
    const float* mach_emb = (const float*)d_in[1];
    const float* agv_emb  = (const float*)d_in[2];
    const float* wait_emb = (const float*)d_in[3];
    const float* pick_w1  = (const float*)d_in[4];
    const float* pick_b1  = (const float*)d_in[5];
    const float* pick_w2  = (const float*)d_in[6];
    const float* pick_b2  = (const float*)d_in[7];
    const float* trans_w1 = (const float*)d_in[8];
    const float* trans_b1 = (const float*)d_in[9];
    const float* trans_w2 = (const float*)d_in[10];
    const float* trans_b2 = (const float*)d_in[11];
    const float* move_w1  = (const float*)d_in[12];
    const float* move_b1  = (const float*)d_in[13];
    const float* move_w2  = (const float*)d_in[14];
    const float* move_b2  = (const float*)d_in[15];
    const int* pick_agv     = (const int*)d_in[16];
    const int* pick_op_from = (const int*)d_in[17];
    const int* pick_op_to   = (const int*)d_in[18];
    const int* pick_machine = (const int*)d_in[19];
    const int* trans_agv    = (const int*)d_in[20];
    const int* trans_mach   = (const int*)d_in[21];
    const int* move_agv     = (const int*)d_in[22];
    const int* move_mach    = (const int*)d_in[23];

    const int nPick  = in_sizes[16];
    const int nTrans = in_sizes[20];
    const int nMove  = in_sizes[22];
    const int nWait  = out_size / 256 - (nPick + nTrans + nMove);

    float* out = (float*)d_out;

    // ---- workspace layout (ushort elements) ----
    // weights (pre-transposed bf16): 1,441,792 elems = 2.9 MB
    unsigned short* ws = (unsigned short*)d_ws;
    unsigned short* pick_w1t  = ws;                  // 512 x 1024
    unsigned short* pick_w2t  = ws + 524288;         // 256 x 512
    unsigned short* trans_w1t = ws + 655360;         // 512 x 512
    unsigned short* trans_w2t = ws + 917504;         // 256 x 512
    unsigned short* move_w1t  = ws + 1048576;        // 512 x 512
    unsigned short* move_w2t  = ws + 1310720;        // 256 x 512
    const size_t W_ELEMS = 1441792;

    // optional bf16 copies of the embedding tables
    const size_t opN   = (size_t)in_sizes[0];
    const size_t machN = (size_t)in_sizes[1];
    const size_t agvN  = (size_t)in_sizes[2];
    unsigned short* op_bf   = ws + W_ELEMS;
    unsigned short* mach_bf = op_bf + opN;
    unsigned short* agv_bf  = mach_bf + machN;
    const bool bigWs = ws_size >= (W_ELEMS + opN + machN + agvN) * sizeof(unsigned short);

    // one-time weight transpose + bf16 downconvert
    cvt_transpose_kernel<<<(1024 * 512 + 255) / 256, 256, 0, stream>>>(pick_w1,  pick_w1t,  1024, 512);
    cvt_transpose_kernel<<<(512  * 256 + 255) / 256, 256, 0, stream>>>(pick_w2,  pick_w2t,  512,  256);
    cvt_transpose_kernel<<<(512  * 512 + 255) / 256, 256, 0, stream>>>(trans_w1, trans_w1t, 512,  512);
    cvt_transpose_kernel<<<(512  * 256 + 255) / 256, 256, 0, stream>>>(trans_w2, trans_w2t, 512,  256);
    cvt_transpose_kernel<<<(512  * 512 + 255) / 256, 256, 0, stream>>>(move_w1,  move_w1t,  512,  512);
    cvt_transpose_kernel<<<(512  * 256 + 255) / 256, 256, 0, stream>>>(move_w2,  move_w2t,  512,  256);

    if (bigWs) {
        cvt_table_kernel<<<((int)(opN / 4)   + 255) / 256, 256, 0, stream>>>(op_emb,   op_bf,   (int)(opN / 4));
        cvt_table_kernel<<<((int)(machN / 4) + 255) / 256, 256, 0, stream>>>(mach_emb, mach_bf, (int)(machN / 4));
        cvt_table_kernel<<<((int)(agvN / 4)  + 255) / 256, 256, 0, stream>>>(agv_emb,  agv_bf,  (int)(agvN / 4));
    }

    // wait rows
    if (nWait > 0) {
        const int n = nWait * 256;
        wait_fill_kernel<<<(n + 255) / 256, 256, 0, stream>>>(wait_emb, out, n);
    }

    if (bigWs) {
        // bf16-table path: A fragments load bf16 directly (no cvt in hot loop)
        mlp2_wmma_kernel<1024, true><<<(nPick + 63) / 64, 256, 0, stream>>>(
            agv_bf, pick_agv, op_bf, pick_op_from, op_bf, pick_op_to, mach_bf, pick_machine,
            pick_w1t, pick_b1, pick_w2t, pick_b2,
            out, (long long)nWait, nPick);
        mlp2_wmma_kernel<512, true><<<(nTrans + 63) / 64, 256, 0, stream>>>(
            agv_bf, trans_agv, mach_bf, trans_mach, nullptr, nullptr, nullptr, nullptr,
            trans_w1t, trans_b1, trans_w2t, trans_b2,
            out, (long long)nWait + nPick, nTrans);
        mlp2_wmma_kernel<512, true><<<(nMove + 63) / 64, 256, 0, stream>>>(
            agv_bf, move_agv, mach_bf, move_mach, nullptr, nullptr, nullptr, nullptr,
            move_w1t, move_b1, move_w2t, move_b2,
            out, (long long)nWait + nPick + nTrans, nMove);
    } else {
        // fallback: gather f32 tables, convert inline
        mlp2_wmma_kernel<1024, false><<<(nPick + 63) / 64, 256, 0, stream>>>(
            agv_emb, pick_agv, op_emb, pick_op_from, op_emb, pick_op_to, mach_emb, pick_machine,
            pick_w1t, pick_b1, pick_w2t, pick_b2,
            out, (long long)nWait, nPick);
        mlp2_wmma_kernel<512, false><<<(nTrans + 63) / 64, 256, 0, stream>>>(
            agv_emb, trans_agv, mach_emb, trans_mach, nullptr, nullptr, nullptr, nullptr,
            trans_w1t, trans_b1, trans_w2t, trans_b2,
            out, (long long)nWait + nPick, nTrans);
        mlp2_wmma_kernel<512, false><<<(nMove + 63) / 64, 256, 0, stream>>>(
            agv_emb, move_agv, mach_emb, move_mach, nullptr, nullptr, nullptr, nullptr,
            move_w1t, move_b1, move_w2t, move_b2,
            out, (long long)nWait + nPick + nTrans, nMove);
    }
}
